// TACTiCSNet_14044543058208
// MI455X (gfx1250) — matrix operand order repaired and checked by reading, not compiled, not run on hardware
//
#include <hip/hip_runtime.h>
#include <hip/hip_bf16.h>

// ---------------------------------------------------------------------------
// TACTiCS-style pipeline for MI455X (gfx1250, wave32, WMMA).
// Dominant cost: two 4096x3000x3000 GEMMs -> v_wmma_f32_16x16x32_f16.
// ---------------------------------------------------------------------------

typedef __attribute__((ext_vector_type(16))) _Float16 v16h;
typedef __attribute__((ext_vector_type(8)))  _Float16 v8h;
typedef __attribute__((ext_vector_type(4)))  _Float16 v4h;
typedef __attribute__((ext_vector_type(8)))  float    v8f;

#define N_CELLS 4096
#define N_GENES 3000
#define H1      64
#define H2      32
#define NC_A    30
#define NC_B    25
#define KNN     20

#define TILE_M 128
#define TILE_N 64
#define TILE_K 32

// ---------------------------------------------------------------------------
// norm_A[r] = sum_c G[r][c]  (zeros -> 1).  One block per row, LDS reduce.
// ---------------------------------------------------------------------------
__global__ __launch_bounds__(256)
void tc_rowsum_kernel(const float* __restrict__ G, float* __restrict__ normA, int n) {
    __shared__ float red[256];
    const int r = blockIdx.x;
    float s = 0.0f;
    for (int c = threadIdx.x; c < n; c += 256) s += G[(size_t)r * n + c];
    red[threadIdx.x] = s;
    __syncthreads();
    for (int off = 128; off > 0; off >>= 1) {
        if (threadIdx.x < off) red[threadIdx.x] += red[threadIdx.x + off];
        __syncthreads();
    }
    if (threadIdx.x == 0) {
        float v = red[0];
        normA[r] = (v == 0.0f) ? 1.0f : v;
    }
}

// norm_B[c] = sum_r G[r][c]  (zeros -> 1).  Thread per column (coalesced rows).
__global__ __launch_bounds__(256)
void tc_colsum_kernel(const float* __restrict__ G, float* __restrict__ normB, int n) {
    const int c = blockIdx.x * blockDim.x + threadIdx.x;
    if (c >= n) return;
    float s = 0.0f;
    for (int r = 0; r < n; ++r) s += G[(size_t)r * n + c];
    normB[c] = (s == 0.0f) ? 1.0f : s;
}

// ---------------------------------------------------------------------------
// C[M,N] = (A[M,K] @ B[K,N]) / norm[N],  B = G or G^T (index swap at staging).
// Block tile 128x64, 8 waves in 4x2; each wave computes 32x32 with four
// accumulators (2 A-frags x 2 B-frags -> 4 WMMA / K-step).  A tile row-major
// in LDS, B tile transposed (sBt[n][k]); all fragment reads are contiguous
// ds_load_b128 pairs.  K-tail iteration peeled out of the hot loop.
// ---------------------------------------------------------------------------
__global__ __launch_bounds__(256)
void tc_impute_gemm_kernel(const float* __restrict__ A,
                           const float* __restrict__ G,     // N_GENES x N_GENES
                           const float* __restrict__ norm,  // [N]
                           float* __restrict__ C,
                           int M, int N, int K, int transB) {
    __shared__ _Float16 sA[TILE_M * TILE_K];    // [m][k]   128x32
    __shared__ _Float16 sBt[TILE_N * TILE_K];   // [n][k]   64x32 (transposed)

    const int tid  = threadIdx.x;
    const int n0   = blockIdx.x * TILE_N;
    const int m0   = blockIdx.y * TILE_M;       // M % TILE_M == 0 -> rows always valid

    const int wave = tid >> 5;        // 0..7
    const int lane = tid & 31;
    const int wm   = wave & 3;        // 32-row slot 0..3
    const int wn   = wave >> 2;       // 32-col slot 0..1
    const int hb   = lane >> 4;       // lane half
    const int l15  = lane & 15;

    const bool fullN = (n0 + TILE_N) <= N;

    v8f acc00 = {}, acc01 = {}, acc10 = {}, acc11 = {};

    // ---- staging helpers -------------------------------------------------
    auto stageA_fast = [&](int k0) {
        // 128x32 f16 = 1024 float4 loads, 4 per thread; aligned (K*4 % 16 == 0).
        #pragma unroll
        for (int e = 0; e < 4; ++e) {
            const int q = e * 256 + tid;          // 0..1023
            const int r = q >> 3;                 // 0..127
            const int c = (q & 7) * 4;            // 0,4,...,28
            const float4 v = *(const float4*)&A[(size_t)(m0 + r) * K + k0 + c];
            v4h hv;
            hv[0] = (_Float16)v.x; hv[1] = (_Float16)v.y;
            hv[2] = (_Float16)v.z; hv[3] = (_Float16)v.w;
            *(v4h*)&sA[r * TILE_K + c] = hv;
        }
    };
    auto stageA_guard = [&](int k0) {
        // Clamped unconditional scalar loads (no exec-mask serialization).
        #pragma unroll
        for (int e = 0; e < 16; ++e) {
            const int idx = e * 256 + tid;        // 0..4095
            const int r = idx >> 5, c = idx & 31;
            const int gk  = k0 + c;
            const int gkc = gk < (K - 1) ? gk : (K - 1);
            const float v = A[(size_t)(m0 + r) * K + gkc];
            sA[idx] = (_Float16)((gk < K) ? v : 0.0f);
        }
    };
    auto stageB_fast = [&](int k0) {
        if (transB) {
            // B[k][n] = G[n][k]: contiguous in k -> float4 along k.
            #pragma unroll
            for (int e = 0; e < 2; ++e) {
                const int q  = e * 256 + tid;     // 0..511
                const int nn = q >> 3;            // 0..63
                const int kk = (q & 7) * 4;       // 0..28
                const float4 v = *(const float4*)&G[(size_t)(n0 + nn) * K + k0 + kk];
                v4h hv;
                hv[0] = (_Float16)v.x; hv[1] = (_Float16)v.y;
                hv[2] = (_Float16)v.z; hv[3] = (_Float16)v.w;
                *(v4h*)&sBt[nn * TILE_K + kk] = hv;
            }
        } else {
            // B[k][n] = G[k][n]: contiguous in n -> float4 along n, scatter
            // 4 scalar f16 stores into the transposed tile.
            #pragma unroll
            for (int e = 0; e < 2; ++e) {
                const int q  = e * 256 + tid;     // 0..511
                const int kk = q >> 4;            // 0..31
                const int nn = (q & 15) * 4;      // 0..60
                const float4 v = *(const float4*)&G[(size_t)(k0 + kk) * N + n0 + nn];
                sBt[(nn + 0) * TILE_K + kk] = (_Float16)v.x;
                sBt[(nn + 1) * TILE_K + kk] = (_Float16)v.y;
                sBt[(nn + 2) * TILE_K + kk] = (_Float16)v.z;
                sBt[(nn + 3) * TILE_K + kk] = (_Float16)v.w;
            }
        }
    };
    auto stageB_guard = [&](int k0) {
        #pragma unroll
        for (int e = 0; e < 8; ++e) {
            const int idx = e * 256 + tid;        // 0..2047
            const int kk = idx >> 6;              // 0..31
            const int nn = idx & 63;              // 0..63
            const int gk = k0 + kk, gn = n0 + nn;
            const int gkc = gk < (K - 1) ? gk : (K - 1);
            const int gnc = gn < (N - 1) ? gn : (N - 1);
            const float v = transB ? G[(size_t)gnc * K + gkc]
                                   : G[(size_t)gkc * N + gnc];
            sBt[nn * TILE_K + kk] = (_Float16)((gk < K && gn < N) ? v : 0.0f);
        }
    };
    // fragments + 4 WMMA for the currently staged tile
    auto do_mma = [&]() {
        const _Float16* arow0 = sA + (wm * 32 + l15) * TILE_K;
        const _Float16* arow1 = arow0 + 16 * TILE_K;
        const v8h a00 = *(const v8h*)(arow0 + 8 * hb);
        const v8h a01 = *(const v8h*)(arow0 + 16 + 8 * hb);
        const v8h a10 = *(const v8h*)(arow1 + 8 * hb);
        const v8h a11 = *(const v8h*)(arow1 + 16 + 8 * hb);
        const v16h af0 = __builtin_shufflevector(a00, a01,
            0, 1, 2, 3, 4, 5, 6, 7, 8, 9, 10, 11, 12, 13, 14, 15);
        const v16h af1 = __builtin_shufflevector(a10, a11,
            0, 1, 2, 3, 4, 5, 6, 7, 8, 9, 10, 11, 12, 13, 14, 15);

        const _Float16* brow0 = sBt + (wn * 32 + l15) * TILE_K;
        const _Float16* brow1 = brow0 + 16 * TILE_K;
        const v8h b0lo = *(const v8h*)(brow0 + 8 * hb);
        const v8h b0hi = *(const v8h*)(brow0 + 16 + 8 * hb);
        const v8h b1lo = *(const v8h*)(brow1 + 8 * hb);
        const v8h b1hi = *(const v8h*)(brow1 + 16 + 8 * hb);
        const v16h bf0 = __builtin_shufflevector(b0lo, b0hi,
            0, 1, 2, 3, 4, 5, 6, 7, 8, 9, 10, 11, 12, 13, 14, 15);
        const v16h bf1 = __builtin_shufflevector(b1lo, b1hi,
            0, 1, 2, 3, 4, 5, 6, 7, 8, 9, 10, 11, 12, 13, 14, 15);

        acc00 = __builtin_amdgcn_wmma_f32_16x16x32_f16(
            false, af0, false, bf0, (short)0, acc00, false, false);
        acc01 = __builtin_amdgcn_wmma_f32_16x16x32_f16(
            false, af0, false, bf1, (short)0, acc01, false, false);
        acc10 = __builtin_amdgcn_wmma_f32_16x16x32_f16(
            false, af1, false, bf0, (short)0, acc10, false, false);
        acc11 = __builtin_amdgcn_wmma_f32_16x16x32_f16(
            false, af1, false, bf1, (short)0, acc11, false, false);
    };

    // ---- interior K-steps (all full); tail peeled below ------------------
    const int kFull = K / TILE_K;
    for (int kt = 0; kt < kFull; ++kt) {
        const int k0 = kt * TILE_K;
        stageA_fast(k0);
        if (fullN) stageB_fast(k0); else stageB_guard(k0);
        __syncthreads();
        do_mma();
        __syncthreads();
    }
    // ---- ragged K tail ---------------------------------------------------
    if (kFull * TILE_K < K) {
        const int k0 = kFull * TILE_K;
        stageA_guard(k0);
        stageB_guard(k0);
        __syncthreads();
        do_mma();
        __syncthreads();
    }

    // C layout: VGPR r -> M = r + 8*half, N = lane&15.  Fused column scaling.
    const int gn0 = n0 + wn * 32 + l15;
    const int gn1 = gn0 + 16;
    const float inv0 = (gn0 < N) ? (1.0f / norm[gn0]) : 0.0f;
    const float inv1 = (gn1 < N) ? (1.0f / norm[gn1]) : 0.0f;
    #pragma unroll
    for (int r = 0; r < 8; ++r) {
        const int gm0 = m0 + wm * 32 + r + 8 * hb;
        const int gm1 = gm0 + 16;
        if (gn0 < N) {
            C[(size_t)gm0 * N + gn0] = acc00[r] * inv0;
            C[(size_t)gm1 * N + gn0] = acc10[r] * inv0;
        }
        if (gn1 < N) {
            C[(size_t)gm0 * N + gn1] = acc01[r] * inv1;
            C[(size_t)gm1 * N + gn1] = acc11[r] * inv1;
        }
    }
}

// ---------------------------------------------------------------------------
// h = relu([first | second] @ W1 + b1), 6000 -> 64.  Block per row,
// thread per output column: W reads coalesce, x reads broadcast.
// ---------------------------------------------------------------------------
__global__ __launch_bounds__(H1)
void tc_mlp1_kernel(const float* __restrict__ xf, const float* __restrict__ xs,
                    const float* __restrict__ W1, const float* __restrict__ b1,
                    float* __restrict__ h) {
    const int i = blockIdx.x;
    const int j = threadIdx.x;            // 0..63
    float acc = b1[j];
    const float* r0 = xf + (size_t)i * N_GENES;
    const float* r1 = xs + (size_t)i * N_GENES;
    for (int k = 0; k < N_GENES; ++k) acc += r0[k] * W1[(size_t)k * H1 + j];
    for (int k = 0; k < N_GENES; ++k) acc += r1[k] * W1[(size_t)(N_GENES + k) * H1 + j];
    h[(size_t)i * H1 + j] = fmaxf(acc, 0.0f);
}

// emb = relu(h @ W2 + b2), 64 -> 32.
__global__ __launch_bounds__(H2)
void tc_mlp2_kernel(const float* __restrict__ h, const float* __restrict__ W2,
                    const float* __restrict__ b2, float* __restrict__ emb) {
    const int i = blockIdx.x;
    const int j = threadIdx.x;            // 0..31
    float acc = b2[j];
    for (int k = 0; k < H1; ++k) acc += h[(size_t)i * H1 + k] * W2[(size_t)k * H2 + j];
    emb[(size_t)i * H2 + j] = fmaxf(acc, 0.0f);
}

// preds = emb @ Wc + bc  (nc = 30 or 25), written straight to d_out.
__global__ __launch_bounds__(32)
void tc_preds_kernel(const float* __restrict__ emb, const float* __restrict__ Wc,
                     const float* __restrict__ bc, float* __restrict__ out, int nc) {
    const int i = blockIdx.x;
    const int j = threadIdx.x;
    if (j >= nc) return;
    float acc = bc[j];
    for (int k = 0; k < H2; ++k) acc += emb[(size_t)i * H2 + k] * Wc[(size_t)k * nc + j];
    out[(size_t)i * nc + j] = acc;
}

// squared L2 norms of embedding rows.
__global__ __launch_bounds__(256)
void tc_sqnorm_kernel(const float* __restrict__ emb, float* __restrict__ s, int n) {
    const int i = blockIdx.x * blockDim.x + threadIdx.x;
    if (i >= n) return;
    float a = 0.0f;
    for (int k = 0; k < H2; ++k) { float v = emb[(size_t)i * H2 + k]; a += v * v; }
    s[i] = a;
}

// ---------------------------------------------------------------------------
// Fused cdist + top-20: per query row, rank db rows by sdb[j] - 2*dot
// (monotone in dist; ||q||^2 constant per row; sqrt monotone -> same order).
// Stable ties (earlier index wins) matches jax.lax.top_k.
// ---------------------------------------------------------------------------
__global__ __launch_bounds__(256)
void tc_topk_kernel(const float* __restrict__ q, const float* __restrict__ db,
                    const float* __restrict__ sdb, int nq, int nd,
                    int* __restrict__ nn) {
    const int i = blockIdx.x * blockDim.x + threadIdx.x;
    if (i >= nq) return;
    float qr[H2];
    for (int k = 0; k < H2; ++k) qr[k] = q[(size_t)i * H2 + k];
    float best[KNN];
    int   bidx[KNN];
    for (int t = 0; t < KNN; ++t) { best[t] = 3.0e38f; bidx[t] = 0; }
    for (int j = 0; j < nd; ++j) {
        float dot = 0.0f;
        for (int k = 0; k < H2; ++k) dot += qr[k] * db[(size_t)j * H2 + k];
        const float score = sdb[j] - 2.0f * dot;
        if (score < best[KNN - 1]) {
            int p = KNN - 1;
            while (p > 0 && score < best[p - 1]) {
                best[p] = best[p - 1]; bidx[p] = bidx[p - 1]; --p;
            }
            best[p] = score; bidx[p] = j;
        }
    }
    for (int t = 0; t < KNN; ++t) nn[(size_t)i * KNN + t] = bidx[t];
}

// ---------------------------------------------------------------------------
// pred = mean over 20 gathered impute rows; accumulate sum((pred - x)^2).
// ---------------------------------------------------------------------------
__global__ __launch_bounds__(256)
void tc_gather_loss_kernel(const int* __restrict__ nn, const float* __restrict__ imp,
                           const float* __restrict__ x, float* __restrict__ acc) {
    __shared__ int   idx[KNN];
    __shared__ float red[256];
    const int i = blockIdx.x;
    if (threadIdx.x < KNN) idx[threadIdx.x] = nn[(size_t)i * KNN + threadIdx.x];
    __syncthreads();
    float local = 0.0f;
    for (int g = threadIdx.x; g < N_GENES; g += 256) {
        float s = 0.0f;
        #pragma unroll
        for (int t = 0; t < KNN; ++t) s += imp[(size_t)idx[t] * N_GENES + g];
        s *= (1.0f / KNN);
        const float d = s - x[(size_t)i * N_GENES + g];
        local += d * d;
    }
    red[threadIdx.x] = local;
    __syncthreads();
    for (int off = 128; off > 0; off >>= 1) {
        if (threadIdx.x < off) red[threadIdx.x] += red[threadIdx.x + off];
        __syncthreads();
    }
    if (threadIdx.x == 0) atomicAdd(acc, red[0]);
}

__global__ void tc_zero_kernel(float* p) { p[0] = 0.0f; }

__global__ void tc_finalize_kernel(const float* __restrict__ acc, float* __restrict__ out) {
    // loss = (sumA + sumB) / (4096*3000)  (both means share the denominator)
    out[0] = acc[0] / ((float)N_CELLS * (float)N_GENES);
}

// ---------------------------------------------------------------------------
// Host-side orchestration (all launches on `stream`; graph-capture safe).
// ---------------------------------------------------------------------------
extern "C" void kernel_launch(void* const* d_in, const int* in_sizes, int n_in,
                              void* d_out, int out_size, void* d_ws, size_t ws_size,
                              hipStream_t stream) {
    (void)in_sizes; (void)n_in; (void)out_size; (void)ws_size;

    const float* x_A  = (const float*)d_in[0];
    const float* x_B  = (const float*)d_in[1];
    const float* G    = (const float*)d_in[2];
    const float* W1   = (const float*)d_in[3];
    const float* b1   = (const float*)d_in[4];
    const float* W2   = (const float*)d_in[5];
    const float* b2   = (const float*)d_in[6];
    const float* WcA  = (const float*)d_in[7];
    const float* bcA  = (const float*)d_in[8];
    const float* WcB  = (const float*)d_in[9];
    const float* bcB  = (const float*)d_in[10];

    float* out = (float*)d_out;
    float* out_predsA = out;                                   // 4096*30
    float* out_predsB = out + (size_t)N_CELLS * NC_A;          // 4096*25
    float* out_loss   = out + (size_t)N_CELLS * (NC_A + NC_B); // 1

    // Workspace carve-up (~102 MB total).
    char* base = (char*)d_ws;
    size_t o = 0;
    auto alloc = [&](size_t bytes) -> char* {
        char* p = base + o;
        o += (bytes + 255) & ~(size_t)255;
        return p;
    };
    float* normA = (float*)alloc((size_t)N_GENES * 4);
    float* normB = (float*)alloc((size_t)N_GENES * 4);
    float* impA  = (float*)alloc((size_t)N_CELLS * N_GENES * 4);
    float* impB  = (float*)alloc((size_t)N_CELLS * N_GENES * 4);
    float* hA    = (float*)alloc((size_t)N_CELLS * H1 * 4);
    float* hB    = (float*)alloc((size_t)N_CELLS * H1 * 4);
    float* embA  = (float*)alloc((size_t)N_CELLS * H2 * 4);
    float* embB  = (float*)alloc((size_t)N_CELLS * H2 * 4);
    float* sA    = (float*)alloc((size_t)N_CELLS * 4);
    float* sB    = (float*)alloc((size_t)N_CELLS * 4);
    int*   nnA   = (int*)  alloc((size_t)N_CELLS * KNN * 4);
    int*   nnB   = (int*)  alloc((size_t)N_CELLS * KNN * 4);
    float* lacc  = (float*)alloc(4);

    // 1) normalizers from gene_dist
    tc_rowsum_kernel<<<N_GENES, 256, 0, stream>>>(G, normA, N_GENES);
    tc_colsum_kernel<<<(N_GENES + 255) / 256, 256, 0, stream>>>(G, normB, N_GENES);

    // 2) imputation GEMMs (WMMA) with fused column normalization
    dim3 gemm_grid((N_GENES + TILE_N - 1) / TILE_N, N_CELLS / TILE_M);
    tc_impute_gemm_kernel<<<gemm_grid, 256, 0, stream>>>(
        x_A, G, normB, impA, N_CELLS, N_GENES, N_GENES, /*transB=*/0);
    tc_impute_gemm_kernel<<<gemm_grid, 256, 0, stream>>>(
        x_B, G, normA, impB, N_CELLS, N_GENES, N_GENES, /*transB=*/1);

    // 3) shared MLP:  x_shared_A = [x_A | impA],  x_shared_B = [impB | x_B]
    tc_mlp1_kernel<<<N_CELLS, H1, 0, stream>>>(x_A, impA, W1, b1, hA);
    tc_mlp1_kernel<<<N_CELLS, H1, 0, stream>>>(impB, x_B, W1, b1, hB);
    tc_mlp2_kernel<<<N_CELLS, H2, 0, stream>>>(hA, W2, b2, embA);
    tc_mlp2_kernel<<<N_CELLS, H2, 0, stream>>>(hB, W2, b2, embB);

    // 4) classifier heads -> d_out
    tc_preds_kernel<<<N_CELLS, 32, 0, stream>>>(embA, WcA, bcA, out_predsA, NC_A);
    tc_preds_kernel<<<N_CELLS, 32, 0, stream>>>(embB, WcB, bcB, out_predsB, NC_B);

    // 5) fused cdist + top-20 neighbor selection (both directions)
    tc_sqnorm_kernel<<<(N_CELLS + 255) / 256, 256, 0, stream>>>(embA, sA, N_CELLS);
    tc_sqnorm_kernel<<<(N_CELLS + 255) / 256, 256, 0, stream>>>(embB, sB, N_CELLS);
    tc_topk_kernel<<<(N_CELLS + 255) / 256, 256, 0, stream>>>(embA, embB, sB, N_CELLS, N_CELLS, nnA);
    tc_topk_kernel<<<(N_CELLS + 255) / 256, 256, 0, stream>>>(embB, embA, sA, N_CELLS, N_CELLS, nnB);

    // 6) embedding-bag mean gather + MSE loss
    tc_zero_kernel<<<1, 1, 0, stream>>>(lacc);
    tc_gather_loss_kernel<<<N_CELLS, 256, 0, stream>>>(nnA, impB, x_A, lacc);
    tc_gather_loss_kernel<<<N_CELLS, 256, 0, stream>>>(nnB, impA, x_B, lacc);
    tc_finalize_kernel<<<1, 1, 0, stream>>>(lacc, out_loss);
}
